// QMEnvironmentEncoder_90941637526151
// MI455X (gfx1250) — matrix operand order repaired and checked by reading, not compiled, not run on hardware
//
#include <hip/hip_runtime.h>
#include <hip/hip_bf16.h>
#include <math.h>

// ---------------- constants (match reference) ----------------
#define DRBF   20
#define CUTF   4.0f
#define DS_    256
#define DV_    128
#define DIN    532        // 256 + 128 + 20 + 128
#define KP1    544        // DIN padded to multiple of 32 (17 chunks)
#define HH     256        // hidden
#define HOUT   128        // output hidden
#define MT     64         // edges per block (four 16-row M tiles)

typedef __attribute__((ext_vector_type(16))) __bf16 v16bf;
typedef __attribute__((ext_vector_type(8)))  __bf16 v8bf;
typedef __attribute__((ext_vector_type(8)))  float  v8f;

union ABfrag { v16bf v; v8bf h[2]; };

__device__ __forceinline__ float silu_f(float x) { return x / (1.0f + __expf(-x)); }

// ---------------- setup: RBF normalization constants ----------------
// rbfC layout: [0..19]=values, [20..39]=mean, [40..59]=inv_std
__global__ void k_rbf_consts(float* rbfC) {
    int j = threadIdx.x;
    if (j >= DRBF) return;
    float step = CUTF / (float)(DRBF - 1);
    float vj = step * (float)j;
    float s = 0.f, s2 = 0.f;
    for (int i = 1; i <= 4000; ++i) {
        float r = (CUTF / 4000.0f) * (float)i;
        float df = (r - vj) / step;
        float b = __expf(-df * df) / 1.12f;
        s += b; s2 += b * b;
    }
    float mean = s / 4000.0f;
    float var = (s2 - 4000.0f * mean * mean) / 3999.0f;   // ddof=1
    rbfC[j] = vj;
    rbfC[DRBF + j] = mean;
    rbfC[2 * DRBF + j] = rsqrtf(var);
}

// ---------------- setup: per-atom vector norms ----------------
__global__ void k_natom(const float* __restrict__ V, float* __restrict__ nat, int n) {
    int i = blockIdx.x * blockDim.x + threadIdx.x;
    if (i < n) {
        float x = V[3 * (size_t)i], y = V[3 * (size_t)i + 1], z = V[3 * (size_t)i + 2];
        nat[i] = sqrtf(x * x + y * y + z * z);
    }
}

// ---------------- setup: tile fp32 weights into WMMA B-fragment bf16 layout ----
// Tile (kc, nt): 32x16 slab of W (row-major Kreal x Ncols), zero-padded in K.
// lane<16:  elems e=0..7 -> K=kc*32+e, e=8..15 -> K=kc*32+8+e,   N=nt*16+lane
// lane>=16: elems e=0..7 -> K=kc*32+8+e, e=8..15 -> K=kc*32+16+e, N=nt*16+lane-16
// Stored contiguously: Wt[tile*512 + lane*16 + e]
__global__ void k_tile_w(const float* __restrict__ W, __bf16* __restrict__ Wt,
                         int Kreal, int Ncols) {
    int tile = blockIdx.x;
    int ntn = Ncols >> 4;
    int kc = tile / ntn, nt = tile % ntn;
    int lane = threadIdx.x;
    int col = nt * 16 + (lane & 15);
    int kbase = kc * 32 + ((lane >> 4) << 3);
    __bf16* dst = Wt + (size_t)tile * 512 + (size_t)lane * 16;
    for (int e = 0; e < 16; ++e) {
        int k = kbase + (e < 8 ? e : e + 8);
        float v = (k < Kreal) ? W[(size_t)k * Ncols + col] : 0.0f;
        dst[e] = (__bf16)v;
    }
}

// ---------------- zero fill ----------------
__global__ void k_zero(float* __restrict__ p, size_t n) {
    size_t i = (size_t)blockIdx.x * blockDim.x + threadIdx.x;
    size_t stride = (size_t)gridDim.x * blockDim.x;
    for (; i < n; i += stride) p[i] = 0.0f;
}

// ---------------- edge kernel: features + LN + 2-layer MLP + scatter ----------
// 256 threads = 8 waves; 64 edges per block = four 16-row M tiles.
// Each wave owns 2 N-tiles and iterates all 4 M-tiles -> each B fragment
// loaded from L2 feeds 4 WMMAs.
__global__ __launch_bounds__(256) void k_edges(
    const float* __restrict__ atom_xyz, const float* __restrict__ probe_xyz,
    const float* __restrict__ cell, const int* __restrict__ edges,
    const float* __restrict__ edisp, const float* __restrict__ Sf,
    const float* __restrict__ Vf, const float* __restrict__ gam,
    const float* __restrict__ bet, const __bf16* __restrict__ W1t,
    const float* __restrict__ b1, const __bf16* __restrict__ W2t,
    const float* __restrict__ b2, const float* __restrict__ nat,
    const float* __restrict__ rbfC, float* __restrict__ m_p, int Etot)
{
    __shared__ __align__(16) __bf16 featA[MT][KP1];     // 69.6 KB
    __shared__ __align__(16) __bf16 h1s[MT][HH];        // 32 KB
    __shared__ int   srcS[MT], dstS[MT];
    __shared__ float dS[MT], envS[MT], rhatS[MT][3];
    __shared__ float redA[MT][4], redB[MT][4], muS[MT], isS[MT];

    const int t = threadIdx.x;
    const int ebase = blockIdx.x * MT;

    // ---- per-edge geometry (64 threads = two waves) ----
    if (t < MT) {
        int e = ebase + t;
        if (e < Etot) {
            int s = edges[2 * (size_t)e], dd = edges[2 * (size_t)e + 1];
            srcS[t] = s; dstS[t] = dd;
            float d0 = edisp[3 * (size_t)e], d1 = edisp[3 * (size_t)e + 1], d2 = edisp[3 * (size_t)e + 2];
            float vec[3];
            for (int j = 0; j < 3; ++j) {
                float dc = d0 * cell[0 * 3 + j] + d1 * cell[1 * 3 + j] + d2 * cell[2 * 3 + j];
                vec[j] = probe_xyz[(size_t)dd * 3 + j] - (atom_xyz[(size_t)s * 3 + j] + dc);
            }
            float d = sqrtf(vec[0] * vec[0] + vec[1] * vec[1] + vec[2] * vec[2]);
            d = fmaxf(d, 1e-8f);
            dS[t] = d;
            for (int j = 0; j < 3; ++j) rhatS[t][j] = vec[j] / d;
            float x = d * 0.25f;
            float x2 = x * x;
            float x5 = x2 * x2 * x;
            float env = 1.0f - 21.0f * x5 + 35.0f * x5 * x - 15.0f * x5 * x * x;
            envS[t] = (d < CUTF) ? env : 0.0f;
        } else {
            srcS[t] = 0; dstS[t] = 0; dS[t] = 1.0f; envS[t] = 0.0f;
            rhatS[t][0] = rhatS[t][1] = rhatS[t][2] = 0.0f;
        }
    }
    __syncthreads();

    // ---- feature fill: 4 threads per edge, store bf16, f32 stats in regs ----
    const int el = t >> 2, li = t & 3;
    const int s = srcS[el];
    float sum = 0.f, sum2 = 0.f;
    for (int j = li; j < DS_; j += 4) {                  // S_final[src]
        float v = Sf[(size_t)s * DS_ + j];
        featA[el][j] = (__bf16)v; sum += v; sum2 += v * v;
    }
    for (int j = li; j < DV_; j += 4) {                  // ||V||[src]
        float v = nat[(size_t)s * DV_ + j];
        featA[el][DS_ + j] = (__bf16)v; sum += v; sum2 += v * v;
    }
    {
        float dd = dS[el];
        const float inv_step = (float)(DRBF - 1) / CUTF;
        for (int j = li; j < DRBF; j += 4) {             // RBF
            float diff = (dd - rbfC[j]) * inv_step;
            float v = (__expf(-diff * diff) / 1.12f - rbfC[DRBF + j]) * rbfC[2 * DRBF + j];
            featA[el][DS_ + DV_ + j] = (__bf16)v; sum += v; sum2 += v * v;
        }
    }
    {
        float r0 = rhatS[el][0], r1 = rhatS[el][1], r2 = rhatS[el][2];
        for (int c = li; c < DV_; c += 4) {              // q = r_hat . V[src]
            const float* vp = Vf + ((size_t)s * DV_ + c) * 3;
            float v = r0 * vp[0] + r1 * vp[1] + r2 * vp[2];
            featA[el][DS_ + DV_ + DRBF + c] = (__bf16)v; sum += v; sum2 += v * v;
        }
    }
    for (int j = DIN + li; j < KP1; j += 4)              // K padding
        featA[el][j] = (__bf16)0.0f;
    redA[el][li] = sum; redB[el][li] = sum2;
    __syncthreads();
    if (li == 0) {
        float S1 = 0.f, S2 = 0.f;
        for (int k = 0; k < 4; ++k) { S1 += redA[el][k]; S2 += redB[el][k]; }
        float mu = S1 / (float)DIN;
        float var = S2 / (float)DIN - mu * mu;
        muS[el] = mu; isS[el] = rsqrtf(var + 1e-5f);
    }
    __syncthreads();
    {   // normalize in place (each element owned by one thread)
        float mu = muS[el], istd = isS[el];
        for (int j = li; j < DIN; j += 4) {
            float v = (float)featA[el][j];
            featA[el][j] = (__bf16)((v - mu) * istd * gam[j] + bet[j]);
        }
    }
    __syncthreads();

    // ---- GEMM 1: (64 x 544) x (544 x 256) via WMMA bf16 ----
    const int lane = t & 31, wave = t >> 5;
    const int rowA = lane & 15, hi = lane >> 4;
    const int nt0 = wave * 2, nt1 = wave * 2 + 1;
    const int colN = lane & 15;
    v8f acc[4][2];                      // acc[mTile][nTile]
    {
        float bias0 = b1[nt0 * 16 + colN];
        float bias1 = b1[nt1 * 16 + colN];
#pragma unroll
        for (int mt = 0; mt < 4; ++mt)
            for (int r = 0; r < 8; ++r) { acc[mt][0][r] = bias0; acc[mt][1][r] = bias1; }
    }
    const v16bf* bp1 = (const v16bf*)W1t;
    for (int kc = 0; kc < KP1 / 32; ++kc) {
        if (kc + 1 < KP1 / 32) {        // prefetch next K-chunk's weight tiles
            __builtin_prefetch(&bp1[(size_t)((kc + 1) * 16 + nt0) * 32 + lane], 0, 1);
            __builtin_prefetch(&bp1[(size_t)((kc + 1) * 16 + nt1) * 32 + lane], 0, 1);
        }
        v16bf b0v = bp1[(size_t)(kc * 16 + nt0) * 32 + lane];
        v16bf b1v = bp1[(size_t)(kc * 16 + nt1) * 32 + lane];
        int koff = kc * 32 + hi * 8;
#pragma unroll
        for (int mt = 0; mt < 4; ++mt) {
            ABfrag a;
            a.h[0] = *(const v8bf*)&featA[mt * 16 + rowA][koff];
            a.h[1] = *(const v8bf*)&featA[mt * 16 + rowA][koff + 16];
            acc[mt][0] = __builtin_amdgcn_wmma_f32_16x16x32_bf16(false, a.v, false, b0v, (short)0, acc[mt][0], false, false);
            acc[mt][1] = __builtin_amdgcn_wmma_f32_16x16x32_bf16(false, a.v, false, b1v, (short)0, acc[mt][1], false, false);
        }
    }
#pragma unroll
    for (int mt = 0; mt < 4; ++mt)       // SiLU -> LDS bf16
        for (int r = 0; r < 8; ++r) {
            int M = mt * 16 + r + hi * 8;
            h1s[M][nt0 * 16 + colN] = (__bf16)silu_f(acc[mt][0][r]);
            h1s[M][nt1 * 16 + colN] = (__bf16)silu_f(acc[mt][1][r]);
        }
    __syncthreads();

    // ---- GEMM 2: (64 x 256) x (256 x 256) ----
    {
        float bias0 = b2[nt0 * 16 + colN];
        float bias1 = b2[nt1 * 16 + colN];
#pragma unroll
        for (int mt = 0; mt < 4; ++mt)
            for (int r = 0; r < 8; ++r) { acc[mt][0][r] = bias0; acc[mt][1][r] = bias1; }
    }
    const v16bf* bp2 = (const v16bf*)W2t;
    for (int kc = 0; kc < HH / 32; ++kc) {
        if (kc + 1 < HH / 32) {
            __builtin_prefetch(&bp2[(size_t)((kc + 1) * 16 + nt0) * 32 + lane], 0, 1);
            __builtin_prefetch(&bp2[(size_t)((kc + 1) * 16 + nt1) * 32 + lane], 0, 1);
        }
        v16bf b0v = bp2[(size_t)(kc * 16 + nt0) * 32 + lane];
        v16bf b1v = bp2[(size_t)(kc * 16 + nt1) * 32 + lane];
        int koff = kc * 32 + hi * 8;
#pragma unroll
        for (int mt = 0; mt < 4; ++mt) {
            ABfrag a;
            a.h[0] = *(const v8bf*)&h1s[mt * 16 + rowA][koff];
            a.h[1] = *(const v8bf*)&h1s[mt * 16 + rowA][koff + 16];
            acc[mt][0] = __builtin_amdgcn_wmma_f32_16x16x32_bf16(false, a.v, false, b0v, (short)0, acc[mt][0], false, false);
            acc[mt][1] = __builtin_amdgcn_wmma_f32_16x16x32_bf16(false, a.v, false, b1v, (short)0, acc[mt][1], false, false);
        }
    }
    // ---- SiLU * env, scatter-add to probes ----
#pragma unroll
    for (int mt = 0; mt < 4; ++mt)
        for (int r = 0; r < 8; ++r) {
            int M = mt * 16 + r + hi * 8;
            float e = envS[M];
            size_t base = (size_t)dstS[M] * HH;
            atomicAdd(&m_p[base + nt0 * 16 + colN], silu_f(acc[mt][0][r]) * e);
            atomicAdd(&m_p[base + nt1 * 16 + colN], silu_f(acc[mt][1][r]) * e);
        }
}

// ---------------- probe kernel: m_p -> silu(m_p@Wo1+bo1) -> @Wo2+bo2 ---------
__global__ __launch_bounds__(256) void k_probes(
    const float* __restrict__ m_p, const __bf16* __restrict__ Wo1t,
    const float* __restrict__ bo1, const float* __restrict__ Wo2,
    const float* __restrict__ bo2, float* __restrict__ rho, int Ptot)
{
    __shared__ __align__(16) float oS[16][HOUT];
    __shared__ float redP[16][16];
    const int t = threadIdx.x;
    const int m0 = blockIdx.x * 16;
    const int lane = t & 31, wave = t >> 5;
    const int rowA = lane & 15, hi = lane >> 4;
    const int nt = wave;                       // 8 waves <-> 8 N-tiles (HOUT=128)

    v8f acc;
    {
        float bias = bo1[nt * 16 + (lane & 15)];
        for (int r = 0; r < 8; ++r) acc[r] = bias;
    }
    int rIdx = m0 + rowA; if (rIdx >= Ptot) rIdx = Ptot - 1;
    const v16bf* bp = (const v16bf*)Wo1t;
    for (int kc = 0; kc < HH / 32; ++kc) {
        ABfrag a;
        const float* src = m_p + (size_t)rIdx * HH + kc * 32 + hi * 8;
        for (int e = 0; e < 8; ++e) {
            a.v[e]     = (__bf16)src[e];
            a.v[8 + e] = (__bf16)src[16 + e];
        }
        v16bf bv = bp[(size_t)(kc * 8 + nt) * 32 + lane];
        acc = __builtin_amdgcn_wmma_f32_16x16x32_bf16(false, a.v, false, bv, (short)0, acc, false, false);
    }
    for (int r = 0; r < 8; ++r) {
        int M = r + hi * 8;
        oS[M][nt * 16 + (lane & 15)] = silu_f(acc[r]);
    }
    __syncthreads();

    const int p = t >> 4, li = t & 15;
    float sloc = 0.f;
    for (int c = li * 8; c < li * 8 + 8; ++c) sloc += oS[p][c] * Wo2[c];
    redP[p][li] = sloc;
    __syncthreads();
    if (li == 0 && (m0 + p) < Ptot) {
        float tot = 0.f;
        for (int k = 0; k < 16; ++k) tot += redP[p][k];
        rho[m0 + p] = tot + bo2[0];
    }
}

// ---------------- launch ----------------
extern "C" void kernel_launch(void* const* d_in, const int* in_sizes, int n_in,
                              void* d_out, int out_size, void* d_ws, size_t ws_size,
                              hipStream_t stream) {
    const float* atom_xyz = (const float*)d_in[0];
    const float* probe_xyz = (const float*)d_in[1];
    const float* cell = (const float*)d_in[2];
    const int*   edges = (const int*)d_in[3];
    const float* edisp = (const float*)d_in[4];
    const float* Sf    = (const float*)d_in[5];
    const float* Vf    = (const float*)d_in[6];
    const float* gam   = (const float*)d_in[7];
    const float* bet   = (const float*)d_in[8];
    const float* W1    = (const float*)d_in[9];
    const float* b1    = (const float*)d_in[10];
    const float* W2    = (const float*)d_in[11];
    const float* b2    = (const float*)d_in[12];
    const float* Wo1   = (const float*)d_in[13];
    const float* bo1   = (const float*)d_in[14];
    const float* Wo2   = (const float*)d_in[15];
    const float* bo2   = (const float*)d_in[16];

    const int N = in_sizes[0] / 3;
    const int P = in_sizes[1] / 3;
    const int E = in_sizes[3] / 2;

    char* ws = (char*)d_ws;
    size_t off = 0;
    float* rbfC = (float*)(ws + off); off += 256;                       // 60 floats
    float* nat  = (float*)(ws + off); off += (size_t)N * DV_ * 4;
    off = (off + 255) & ~(size_t)255;
    __bf16* W1t  = (__bf16*)(ws + off); off += (size_t)(KP1 / 32) * (HH / 16) * 512 * 2;
    __bf16* W2t  = (__bf16*)(ws + off); off += (size_t)(HH / 32) * (HH / 16) * 512 * 2;
    __bf16* Wo1t = (__bf16*)(ws + off); off += (size_t)(HH / 32) * (HOUT / 16) * 512 * 2;
    off = (off + 255) & ~(size_t)255;
    float* m_p = (float*)(ws + off); off += (size_t)P * HH * 4;

    // setup (cheap, deterministic, re-done each call)
    k_rbf_consts<<<1, 32, 0, stream>>>(rbfC);
    int natN = N * DV_;
    k_natom<<<(natN + 255) / 256, 256, 0, stream>>>(Vf, nat, natN);
    k_tile_w<<<(KP1 / 32) * (HH / 16),   32, 0, stream>>>(W1,  W1t,  DIN, HH);
    k_tile_w<<<(HH  / 32) * (HH / 16),   32, 0, stream>>>(W2,  W2t,  HH,  HH);
    k_tile_w<<<(HH  / 32) * (HOUT / 16), 32, 0, stream>>>(Wo1, Wo1t, HH,  HOUT);
    k_zero<<<2048, 256, 0, stream>>>(m_p, (size_t)P * HH);

    // main pipeline
    k_edges<<<(E + MT - 1) / MT, 256, 0, stream>>>(
        atom_xyz, probe_xyz, cell, edges, edisp, Sf, Vf, gam, bet,
        W1t, b1, W2t, b2, nat, rbfC, m_p, E);
    k_probes<<<(P + 15) / 16, 256, 0, stream>>>(
        m_p, Wo1t, bo1, Wo2, bo2, (float*)d_out, P);
}
